// Link_Prediction_79096117723242
// MI455X (gfx1250) — matrix operand order repaired and checked by reading, not compiled, not run on hardware
//
#include <hip/hip_runtime.h>
#include <hip/hip_bf16.h>

typedef __attribute__((ext_vector_type(16))) _Float16 v16h;
typedef __attribute__((ext_vector_type(8)))  float    v8f;

__device__ __forceinline__ void atomic_add_f32(float* p, float v) {
    __hip_atomic_fetch_add(p, v, __ATOMIC_RELAXED, __HIP_MEMORY_SCOPE_AGENT);
}

// ---------------------------------------------------------------------------
// Pack W[128,HOUT] (f32, row-major) into B-fragment-native f16 layout:
// Wp[(((tile*4 + kb)*32 + lane)*16) + i] = W[(kb*32 + (lane>>4)*16 + i)*HOUT
//                                            + tile*16 + (lane&15)]
// so each lane's 16-half fragment is one aligned 32-byte contiguous load.
// ---------------------------------------------------------------------------
template <int HOUT>
__global__ void pack_w_kernel(const float* __restrict__ W, _Float16* __restrict__ Wp) {
    const int j = blockIdx.x * blockDim.x + threadIdx.x;
    if (j >= 128 * HOUT) return;
    const int i    = j & 15;
    const int lane = (j >> 4) & 31;
    const int kb   = (j >> 9) & 3;
    const int tile = j >> 11;
    const int k    = kb * 32 + (lane >> 4) * 16 + i;
    const int col  = tile * 16 + (lane & 15);
    Wp[j] = (_Float16)W[k * HOUT + col];
}

// ---------------------------------------------------------------------------
// Degree (with self loop) -> dinv = rsqrt(deg), computed in place.
// ---------------------------------------------------------------------------
__global__ void deg_init_kernel(float* __restrict__ deg, int n) {
    int i = blockIdx.x * blockDim.x + threadIdx.x;
    if (i < n) deg[i] = 1.0f;   // self loop
}

__global__ void deg_count_kernel(float* __restrict__ deg, const int* __restrict__ dst, int e) {
    int i = blockIdx.x * blockDim.x + threadIdx.x;
    if (i < e) atomic_add_f32(&deg[dst[i]], 1.0f);
}

__global__ void deg_rsqrt_kernel(float* __restrict__ deg, int n) {
    int i = blockIdx.x * blockDim.x + threadIdx.x;
    if (i < n) deg[i] = rsqrtf(deg[i]);
}

// ---------------------------------------------------------------------------
// WMMA GEMM: Y[N,HOUT] = X[N,128] @ W[128,HOUT].  X f32 staged as f16 in LDS
// (shared by all waves of the block); W pre-packed in fragment order.
// One wave per 16x16 output tile, 4 K-steps of 32.  N % 16 == 0 so EXEC is
// all-ones everywhere (WMMA requirement).
// ---------------------------------------------------------------------------
template <int HOUT>
__global__ void __launch_bounds__(32 * (HOUT / 16))
gemm_wmma_kernel(const float* __restrict__ X, const _Float16* __restrict__ Wp,
                 float* __restrict__ Y) {
    __shared__ _Float16 sA[16 * 128];
    const int rowbase = blockIdx.x * 16;

    // Stage 16x128 f32 tile of X into LDS as f16 (coalesced, converted in flight).
    for (int idx = threadIdx.x; idx < 16 * 128; idx += blockDim.x)
        sA[idx] = (_Float16)X[(size_t)rowbase * 128 + idx];
    __syncthreads();

    const int lane = threadIdx.x & 31;
    const int wave = threadIdx.x >> 5;
    const int m    = lane & 15;     // A row within tile
    const int ksel = lane >> 4;     // half-wave selector
    const int ncol = wave * 16 + (lane & 15);  // output column

    v8f c = {};
#pragma unroll
    for (int kb = 0; kb < 4; ++kb) {
        const int kbase = kb * 32;
        // A fragment: lanes 0-15 hold K 0-15, lanes 16-31 hold K 16-31
        // (contiguous, matching packed W B fragment).
        v16h a;
#pragma unroll
        for (int i = 0; i < 16; ++i) {
            const int kl = ksel * 16 + i;
            a[i] = sA[m * 128 + kbase + kl];
        }
        // B fragment: one aligned 32-byte contiguous load from packed W.
        const v16h b = *(const v16h*)&Wp[(((size_t)wave * 4 + kb) * 32 + lane) * 16];

        c = __builtin_amdgcn_wmma_f32_16x16x32_f16(
                /*neg_a=*/false, a, /*neg_b=*/false, b,
                /*c_mod=*/(short)0, c, /*reuse_a=*/false, /*reuse_b=*/false);
    }

    // D layout: VGPR r -> M = r + 8*(lane>>4), N = lane&15.
#pragma unroll
    for (int r = 0; r < 8; ++r) {
        const int row = rowbase + r + ksel * 8;
        Y[(size_t)row * HOUT + ncol] = c[r];
    }
}

// ---------------------------------------------------------------------------
// Self-loop init: out[i,f] = in[i,f] * dinv[i]^2  (also fully initializes out).
// ---------------------------------------------------------------------------
template <int H>
__global__ void agg_init_kernel(const float* __restrict__ hin, float* __restrict__ hout,
                                const float* __restrict__ dinv, int total) {
    constexpr int SH = (H == 128) ? 7 : 6;
    int i = blockIdx.x * blockDim.x + threadIdx.x;
    if (i >= total) return;
    const float di = dinv[i >> SH];
    hout[i] = hin[i] * di * di;
}

// ---------------------------------------------------------------------------
// Edge scatter-add: wave per edge, lane handles H/32 contiguous features.
// out[dst] += in[src] * dinv[src]*dinv[dst] via global f32 atomics (L2-resident).
// ---------------------------------------------------------------------------
template <int H>
__global__ void agg_edges_kernel(const float* __restrict__ hin, float* __restrict__ hout,
                                 const float* __restrict__ dinv,
                                 const int* __restrict__ src, const int* __restrict__ dst,
                                 int e) {
    constexpr int F = H / 32;
    const int wave = blockIdx.x * (blockDim.x >> 5) + (threadIdx.x >> 5);
    const int lane = threadIdx.x & 31;
    if (wave >= e) return;
    const int s = src[wave];
    const int d = dst[wave];
    const float norm = dinv[s] * dinv[d];
    const float* hs = hin + (size_t)s * H + lane * F;
    float* ho       = hout + (size_t)d * H + lane * F;
    if constexpr (F == 4) {
        const float4 v = *(const float4*)hs;   // global_load_b128
        atomic_add_f32(ho + 0, v.x * norm);
        atomic_add_f32(ho + 1, v.y * norm);
        atomic_add_f32(ho + 2, v.z * norm);
        atomic_add_f32(ho + 3, v.w * norm);
    } else {
        const float2 v = *(const float2*)hs;   // global_load_b64
        atomic_add_f32(ho + 0, v.x * norm);
        atomic_add_f32(ho + 1, v.y * norm);
    }
}

// ---------------------------------------------------------------------------
// Bias (+ optional ReLU), in place.
// ---------------------------------------------------------------------------
template <int H, bool RELU>
__global__ void bias_act_kernel(float* __restrict__ Y, const float* __restrict__ bias, int total) {
    int i = blockIdx.x * blockDim.x + threadIdx.x;
    if (i >= total) return;
    float v = Y[i] + bias[i & (H - 1)];
    Y[i] = RELU ? fmaxf(v, 0.0f) : v;
}

// ---------------------------------------------------------------------------
// Decode: wave per candidate link, 64-wide dot via 2 floats/lane + shuffle tree.
// ---------------------------------------------------------------------------
__global__ void decode_kernel(const float* __restrict__ z, const int* __restrict__ eli,
                              float* __restrict__ out, int el) {
    const int wave = blockIdx.x * (blockDim.x >> 5) + (threadIdx.x >> 5);
    const int lane = threadIdx.x & 31;
    if (wave >= el) return;
    const int a = eli[wave];
    const int b = eli[el + wave];
    const float* za = z + (size_t)a * 64;
    const float* zb = z + (size_t)b * 64;
    float p = za[lane] * zb[lane] + za[lane + 32] * zb[lane + 32];
#pragma unroll
    for (int off = 16; off > 0; off >>= 1)
        p += __shfl_down(p, off, 32);
    if (lane == 0) out[wave] = p;
}

// ---------------------------------------------------------------------------
extern "C" void kernel_launch(void* const* d_in, const int* in_sizes, int n_in,
                              void* d_out, int out_size, void* d_ws, size_t ws_size,
                              hipStream_t stream) {
    const float* x   = (const float*)d_in[0];   // [N,128]
    const float* W1  = (const float*)d_in[1];   // [128,128]
    const float* b1  = (const float*)d_in[2];   // [128]
    const float* W2  = (const float*)d_in[3];   // [128,64]
    const float* b2  = (const float*)d_in[4];   // [64]
    const int*   ei  = (const int*)d_in[5];     // [2,E]
    const int*   eli = (const int*)d_in[6];     // [2,EL]

    const int N  = in_sizes[0] / 128;
    const int E  = in_sizes[5] / 2;
    const int EL = in_sizes[6] / 2;

    const int* src = ei;
    const int* dst = ei + E;

    // Workspace carve (every region we read is fully written before use).
    float* deg = (float*)d_ws;                              // N (becomes dinv)
    float* h0  = deg + (((size_t)N + 127) & ~(size_t)127);  // N*128 : x@W1
    float* h   = h0 + (size_t)N * 128;                      // N*128 : layer-1 agg
    float* z0  = h  + (size_t)N * 128;                      // N*64  : h@W2
    float* zb  = z0 + (size_t)N * 64;                       // N*64  : layer-2 agg
    _Float16* W1p = (_Float16*)(zb + (size_t)N * 64);       // packed fragments
    _Float16* W2p = W1p + 128 * 128;

    const int T = 256;

    // Weight packing + degree normalization.
    pack_w_kernel<128><<<(128 * 128 + T - 1) / T, T, 0, stream>>>(W1, W1p);
    pack_w_kernel<64><<<(128 * 64 + T - 1) / T, T, 0, stream>>>(W2, W2p);
    deg_init_kernel<<<(N + T - 1) / T, T, 0, stream>>>(deg, N);
    deg_count_kernel<<<(E + T - 1) / T, T, 0, stream>>>(deg, dst, E);
    deg_rsqrt_kernel<<<(N + T - 1) / T, T, 0, stream>>>(deg, N);

    // Layer 1: h0 = x @ W1 (WMMA), aggregate, +b1, ReLU.
    gemm_wmma_kernel<128><<<N / 16, 256, 0, stream>>>(x, W1p, h0);
    agg_init_kernel<128><<<(N * 128 + T - 1) / T, T, 0, stream>>>(h0, h, deg, N * 128);
    agg_edges_kernel<128><<<(E + 7) / 8, 256, 0, stream>>>(h0, h, deg, src, dst, E);
    bias_act_kernel<128, true><<<(N * 128 + T - 1) / T, T, 0, stream>>>(h, b1, N * 128);

    // Layer 2: z0 = h @ W2 (WMMA), aggregate, +b2.
    gemm_wmma_kernel<64><<<N / 16, 128, 0, stream>>>(h, W2p, z0);
    agg_init_kernel<64><<<(N * 64 + T - 1) / T, T, 0, stream>>>(z0, zb, deg, N * 64);
    agg_edges_kernel<64><<<(E + 7) / 8, 256, 0, stream>>>(z0, zb, deg, src, dst, E);
    bias_act_kernel<64, false><<<(N * 64 + T - 1) / T, T, 0, stream>>>(zb, b2, N * 64);

    // Decode candidate links.
    decode_kernel<<<(EL + 7) / 8, 256, 0, stream>>>(zb, eli, (float*)d_out, EL);
}
